// CrossAttention_80659485819410
// MI455X (gfx1250) — matrix operand order repaired and checked by reading, not compiled, not run on hardware
//
#include <hip/hip_runtime.h>
#include <hip/hip_bf16.h>

#define B_  2
#define N_  2048
#define M_  2048
#define DIM_ 1024
#define H_  16
#define DH_ 64

typedef __attribute__((ext_vector_type(16))) __bf16 v16bf;
typedef __attribute__((ext_vector_type(8)))  float  v8f;

union FragB16 {
    v16bf v;
    unsigned short u[16];
    uint4 q[2];
};

__device__ __forceinline__ unsigned short f2bf(float f) {
    unsigned int u = __float_as_uint(f);
    unsigned int r = u + 0x7FFFu + ((u >> 16) & 1u);
    return (unsigned short)(r >> 16);
}

#define WMMA_BF16(a, b, c) \
    __builtin_amdgcn_wmma_f32_16x16x32_bf16(false, (a), false, (b), (short)0, (c), false, false)

// CDNA5 async global->LDS copy (ASYNCcnt-tracked, bypasses VGPRs).
// Generic LDS pointer: addr[31:0] is the LDS byte address (aperture rule).
__device__ __forceinline__ void async_load_b128(const void* g, void* l) {
    unsigned lds = (unsigned)(unsigned long long)l;
    asm volatile("global_load_async_to_lds_b128 %0, %1, off"
                 :: "v"(lds), "v"(g) : "memory");
}
__device__ __forceinline__ void wait_async0() {
    asm volatile("s_wait_asynccnt 0" ::: "memory");
}

// A-fragment: lane half 0 -> K {k0+0..7, k0+16..23}; half 1 -> K {k0+8..15, k0+24..31}
__device__ __forceinline__ FragB16 load_a_frag(const unsigned short* base, int row,
                                               int stride, int k0, int half) {
    FragB16 f;
    const unsigned short* p = base + (size_t)row * stride + k0 + half * 8;
    f.q[0] = *(const uint4*)p;
    f.q[1] = *(const uint4*)(p + 16);
    return f;
}

// B-fragment from [col][k] storage: lane half 0 -> K {k0+0..7, k0+16..23}; half 1 -> K {k0+8..15, k0+24..31}
__device__ __forceinline__ FragB16 load_b_frag(const unsigned short* base, int col,
                                               int stride, int k0, int half) {
    FragB16 f;
    const unsigned short* p = base + (size_t)col * stride + k0 + half * 8;
    f.q[0] = *(const uint4*)p;
    f.q[1] = *(const uint4*)(p + 16);
    return f;
}

// ---------------------------------------------------------------------------
// Tiled GEMM, bf16 WMMA, f32 accumulate. Block tile 128x128, K step 64.
// 256 threads = 8 waves, each wave a 32x64 sub-tile (2x4 WMMA tiles, 16 WMMA
// per K step). LDS: A[128x64] + B(transposed)[128x64], stride 72 (16B-aligned,
// bank-skewed).
// MODE 0: C = (A_f32 @ W) * 0.125 -> bf16 q stored [b,h,n,d]
// MODE 1: C =  A_f32 @ W          -> bf16 k,v stored [b,h,m,d]
// MODE 2: C =  A_bf16 @ W + bias  -> f32 output [row][col] (A staged async)
// ---------------------------------------------------------------------------
template <int MODE>
__global__ __launch_bounds__(256) void gemm_bf16_kernel(
    const float* __restrict__ Af, const unsigned short* __restrict__ Ab,
    const float* __restrict__ W, const float* __restrict__ bias,
    unsigned short* __restrict__ out0, unsigned short* __restrict__ out1,
    float* __restrict__ outf, int K, int Ncols) {
    __shared__ unsigned short As[128 * 72];   // [row][k], stride 72
    __shared__ unsigned short Bs[128 * 72];   // [col][k] (transposed), stride 72

    const int tid  = threadIdx.x;
    const int wave = tid >> 5;
    const int lane = tid & 31;
    const int half = lane >> 4;
    const int l15  = lane & 15;
    const int wr   = wave >> 1;   // 0..3 -> 32-row strip
    const int wc   = wave & 1;    // 0..1 -> 64-col strip
    const int row0 = blockIdx.y * 128;
    const int col0 = blockIdx.x * 128;

    v8f c[2][4];
#pragma unroll
    for (int mt = 0; mt < 2; ++mt)
#pragma unroll
        for (int nt = 0; nt < 4; ++nt)
#pragma unroll
            for (int i = 0; i < 8; ++i) c[mt][nt][i] = 0.0f;

    for (int k0 = 0; k0 < K; k0 += 64) {
        // prefetch next K tile into L2 while this one is staged/consumed
        if (k0 + 64 < K) {
            if (MODE != 2)
                __builtin_prefetch(Af + (size_t)(row0 + (tid >> 1)) * K + k0 + 64 + (tid & 1) * 32, 0, 0);
            __builtin_prefetch(W + (size_t)(k0 + 64 + (tid >> 2)) * Ncols + col0 + (tid & 3) * 32, 0, 0);
        }
        __syncthreads();
        // ---- stage A tile 128x64 into LDS ----
        if (MODE == 2) {
#pragma unroll
            for (int p = 0; p < 4; ++p) {
                int idx = tid + p * 256;           // 1024 x 8 bf16, async DMA
                int r = idx >> 3, kq = (idx & 7) * 8;
                async_load_b128(Ab + (size_t)(row0 + r) * K + k0 + kq, As + r * 72 + kq);
            }
        } else {
#pragma unroll
            for (int p = 0; p < 8; ++p) {
                int idx = tid + p * 256;           // 2048 x float4
                int r = idx >> 4, kq = (idx & 15) * 4;
                float4 d = *(const float4*)(Af + (size_t)(row0 + r) * K + k0 + kq);
                As[r * 72 + kq + 0] = f2bf(d.x);
                As[r * 72 + kq + 1] = f2bf(d.y);
                As[r * 72 + kq + 2] = f2bf(d.z);
                As[r * 72 + kq + 3] = f2bf(d.w);
            }
        }
        // ---- stage W tile 64x128 transposed into LDS as [col][k] ----
#pragma unroll
        for (int p = 0; p < 8; ++p) {
            int idx = tid + p * 256;
            int kr = idx >> 5, nq = (idx & 31) * 4;
            float4 d = *(const float4*)(W + (size_t)(k0 + kr) * Ncols + col0 + nq);
            Bs[(nq + 0) * 72 + kr] = f2bf(d.x);
            Bs[(nq + 1) * 72 + kr] = f2bf(d.y);
            Bs[(nq + 2) * 72 + kr] = f2bf(d.z);
            Bs[(nq + 3) * 72 + kr] = f2bf(d.w);
        }
        if (MODE == 2) wait_async0();
        __syncthreads();

#pragma unroll
        for (int ksub = 0; ksub < 64; ksub += 32) {
            FragB16 bfrag[4];
#pragma unroll
            for (int nt = 0; nt < 4; ++nt)
                bfrag[nt] = load_b_frag(Bs, wc * 64 + nt * 16 + l15, 72, ksub, half);
#pragma unroll
            for (int mt = 0; mt < 2; ++mt) {
                FragB16 afrag = load_a_frag(As, wr * 32 + mt * 16 + l15, 72, ksub, half);
#pragma unroll
                for (int nt = 0; nt < 4; ++nt)
                    c[mt][nt] = WMMA_BF16(afrag.v, bfrag[nt].v, c[mt][nt]);
            }
        }
    }

    // ---- store ----
#pragma unroll
    for (int mt = 0; mt < 2; ++mt) {
#pragma unroll
        for (int nt = 0; nt < 4; ++nt) {
#pragma unroll
            for (int r = 0; r < 8; ++r) {
                int m  = r + half * 8;
                int gr = row0 + wr * 32 + mt * 16 + m;
                int gc = col0 + wc * 64 + nt * 16 + l15;
                float val = c[mt][nt][r];
                if (MODE == 0) {
                    int bb = gr >> 11, nn = gr & (N_ - 1);
                    int h = gc >> 6, d = gc & 63;
                    out0[((((size_t)bb * H_ + h) * N_) + nn) * DH_ + d] = f2bf(val * 0.125f);
                } else if (MODE == 1) {
                    int bb = gr >> 11, nn = gr & (M_ - 1);
                    unsigned short* dst = out0;
                    int cc = gc;
                    if (cc >= DIM_) { dst = out1; cc -= DIM_; }
                    int h = cc >> 6, d = cc & 63;
                    dst[((((size_t)bb * H_ + h) * M_) + nn) * DH_ + d] = f2bf(val);
                } else {
                    outf[(size_t)gr * DIM_ + gc] = val + bias[gc];
                }
            }
        }
    }
}

// ---------------------------------------------------------------------------
// Flash attention: one block = (b,h) x 64-query tile; 4 waves, 16 rows each.
// Causal (M==N) => only key chunks jc <= qt; diagonal chunk element-masked.
// K chunk staged via async global->LDS DMA; V staged transposed manually.
// ---------------------------------------------------------------------------
__global__ __launch_bounds__(128) void attn_kernel(
    const unsigned short* __restrict__ qws, const unsigned short* __restrict__ kws,
    const unsigned short* __restrict__ vws, const int* __restrict__ maskp,
    unsigned short* __restrict__ attn_out) {
    __shared__ unsigned short Ks[64 * 80];       // [key][d]
    __shared__ unsigned short Vt[64 * 80];       // [d][key]
    __shared__ unsigned short Ps[4 * 16 * 80];   // per-wave P tile [m][key]
    __shared__ int smask[64];

    const int tid  = threadIdx.x;
    const int wave = tid >> 5;
    const int lane = tid & 31;
    const int half = lane >> 4;
    const int l15  = lane & 15;
    const int qt   = blockIdx.x;       // query tile (64 rows)
    const int bh   = blockIdx.y;       // b*H + h
    const int b    = bh >> 4;
    const int h    = bh & 15;

    const size_t headoff = (size_t)bh * M_ * DH_;
    const unsigned short* qh = qws + headoff;
    const unsigned short* kh = kws + headoff;
    const unsigned short* vh = vws + headoff;
    const int qrow0 = qt * 64 + wave * 16;

    // Q fragments (A-layout), loaded once straight from global bf16
    FragB16 aq[2];
#pragma unroll
    for (int ks = 0; ks < 2; ++ks) {
        const unsigned short* p = qh + (size_t)(qrow0 + l15) * DH_ + ks * 32 + half * 8;
        aq[ks].q[0] = *(const uint4*)p;
        aq[ks].q[1] = *(const uint4*)(p + 16);
    }

    v8f o[4];
#pragma unroll
    for (int nt = 0; nt < 4; ++nt)
#pragma unroll
        for (int i = 0; i < 8; ++i) o[nt][i] = 0.0f;
    float mi[8], li[8];
#pragma unroll
    for (int r = 0; r < 8; ++r) { mi[r] = -1.0e30f; li[r] = 0.0f; }

    unsigned short* Pw = Ps + wave * 16 * 80;

    for (int jc = 0; jc <= qt; ++jc) {
        // prefetch next chunk's K/V lines into L2
        if (jc < qt) {
            const unsigned short* nk = kh + (size_t)(jc + 1) * 64 * DH_;
            const unsigned short* nv = vh + (size_t)(jc + 1) * 64 * DH_;
            const unsigned short* pp = (tid < 64) ? (nk + tid * DH_) : (nv + (tid - 64) * DH_);
            __builtin_prefetch(pp, 0, 0);
        }
        __syncthreads();
        const unsigned short* kc = kh + (size_t)jc * 64 * DH_;
        const unsigned short* vc = vh + (size_t)jc * 64 * DH_;
#pragma unroll
        for (int p = 0; p < 4; ++p) {
            int idx = tid + p * 128;               // 512 x 8 bf16
            int key = idx >> 3, dq = (idx & 7) * 8;
            // K: straight copy -> async DMA to LDS
            async_load_b128(kc + (size_t)key * DH_ + dq, Ks + key * 80 + dq);
            // V: transposed staging (manual)
            uint4 dv = *(const uint4*)(vc + (size_t)key * DH_ + dq);
            const unsigned short* pv = (const unsigned short*)&dv;
#pragma unroll
            for (int i = 0; i < 8; ++i) Vt[(dq + i) * 80 + key] = pv[i];
        }
        if (tid < 64) smask[tid] = maskp[b * M_ + jc * 64 + tid];
        wait_async0();
        __syncthreads();

        // S = Q K^T  (16 x 64 per wave)
        v8f s[4];
#pragma unroll
        for (int nt = 0; nt < 4; ++nt)
#pragma unroll
            for (int i = 0; i < 8; ++i) s[nt][i] = 0.0f;
#pragma unroll
        for (int ks = 0; ks < 2; ++ks) {
#pragma unroll
            for (int nt = 0; nt < 4; ++nt) {
                FragB16 bk = load_b_frag(Ks, nt * 16 + l15, 80, ks * 32, half);
                s[nt] = WMMA_BF16(aq[ks].v, bk.v, s[nt]);
            }
        }

        // masking: padding + causal
#pragma unroll
        for (int nt = 0; nt < 4; ++nt) {
            int keyl = nt * 16 + l15;
            int keyg = jc * 64 + keyl;
            bool pm = (smask[keyl] != 0);
#pragma unroll
            for (int r = 0; r < 8; ++r) {
                int qg = qrow0 + r + half * 8;
                if (!pm || keyg > qg) s[nt][r] = -3.0e37f;
            }
        }

        // online softmax: row max / rescale / exp / row sum
        float mnew[8], alpha[8], rsum[8];
#pragma unroll
        for (int r = 0; r < 8; ++r) {
            float mx = s[0][r];
#pragma unroll
            for (int nt = 1; nt < 4; ++nt) mx = fmaxf(mx, s[nt][r]);
#pragma unroll
            for (int off = 8; off >= 1; off >>= 1) mx = fmaxf(mx, __shfl_xor(mx, off, 16));
            mnew[r]  = fmaxf(mi[r], mx);
            alpha[r] = __expf(mi[r] - mnew[r]);
            rsum[r]  = 0.0f;
        }
#pragma unroll
        for (int nt = 0; nt < 4; ++nt) {
#pragma unroll
            for (int r = 0; r < 8; ++r) {
                float pv = __expf(s[nt][r] - mnew[r]);
                rsum[r] += pv;
                Pw[(r + half * 8) * 80 + nt * 16 + l15] = f2bf(pv);
            }
        }
#pragma unroll
        for (int r = 0; r < 8; ++r) {
            float sm = rsum[r];
#pragma unroll
            for (int off = 8; off >= 1; off >>= 1) sm += __shfl_xor(sm, off, 16);
            li[r] = li[r] * alpha[r] + sm;
            mi[r] = mnew[r];
#pragma unroll
            for (int nt = 0; nt < 4; ++nt) o[nt][r] *= alpha[r];
        }
        __syncthreads();   // make P visible for fragment reloads

        // O += P V
#pragma unroll
        for (int ks = 0; ks < 2; ++ks) {
            FragB16 ap = load_a_frag(Pw, l15, 80, ks * 32, half);
#pragma unroll
            for (int nt = 0; nt < 4; ++nt) {
                FragB16 bv = load_b_frag(Vt, nt * 16 + l15, 80, ks * 32, half);
                o[nt] = WMMA_BF16(ap.v, bv.v, o[nt]);
            }
        }
    }

    // epilogue: normalize and store attn output [b*N + q][h*64 + d] bf16
#pragma unroll
    for (int r = 0; r < 8; ++r) {
        float inv = 1.0f / fmaxf(li[r], 1.0e-30f);
        int qg = qrow0 + r + half * 8;
        size_t rowb = ((size_t)b * N_ + qg) * (H_ * DH_);
#pragma unroll
        for (int nt = 0; nt < 4; ++nt)
            attn_out[rowb + h * DH_ + nt * 16 + l15] = f2bf(o[nt][r] * inv);
    }
}

extern "C" void kernel_launch(void* const* d_in, const int* in_sizes, int n_in,
                              void* d_out, int out_size, void* d_ws, size_t ws_size,
                              hipStream_t stream) {
    const float* x    = (const float*)d_in[0];
    const float* cond = (const float*)d_in[1];
    const int*   mask = (const int*)d_in[2];
    const float* Wq   = (const float*)d_in[3];
    const float* Wkv  = (const float*)d_in[4];
    const float* Wo   = (const float*)d_in[5];
    const float* bo   = (const float*)d_in[6];
    float* out        = (float*)d_out;

    const size_t HEAD_ELEMS = (size_t)B_ * H_ * N_ * DH_;   // 4M bf16 each
    unsigned short* qws = (unsigned short*)d_ws;
    unsigned short* kws = qws + HEAD_ELEMS;
    unsigned short* vws = kws + HEAD_ELEMS;
    unsigned short* aws = vws + HEAD_ELEMS;

    dim3 blk(256);
    // q = (x @ Wq) * scale   : 4096 x 1024
    gemm_bf16_kernel<0><<<dim3(DIM_ / 128, (B_ * N_) / 128), blk, 0, stream>>>(
        x, nullptr, Wq, nullptr, qws, nullptr, nullptr, DIM_, DIM_);
    // kv = cond @ Wkv        : 4096 x 2048
    gemm_bf16_kernel<1><<<dim3((2 * DIM_) / 128, (B_ * M_) / 128), blk, 0, stream>>>(
        cond, nullptr, Wkv, nullptr, kws, vws, nullptr, DIM_, 2 * DIM_);
    // attention
    attn_kernel<<<dim3(N_ / 64, B_ * H_), dim3(128), 0, stream>>>(qws, kws, vws, mask, aws);
    // out = attn @ Wo + bo   : 4096 x 1024, f32
    gemm_bf16_kernel<2><<<dim3(DIM_ / 128, (B_ * N_) / 128), blk, 0, stream>>>(
        nullptr, aws, Wo, bo, nullptr, nullptr, out, DIM_, DIM_);
}
